// DHSRNNCell_86474871538252
// MI455X (gfx1250) — matrix operand order repaired and checked by reading, not compiled, not run on hardware
//
#include <hip/hip_runtime.h>

// Problem dims (from reference): B=8192, IN=1024, H=1024, K=4 branches, BR=512.
#define B_DIM   8192
#define IN_DIM  1024
#define H_DIM   1024
#define K_BR    4
#define BR_DIM  512
#define KD      2048   // fused inner dim = K_BR * BR_DIM

// GEMM tiling
#define BM   128
#define BN   128
#define BK   64
#define LDSS 72        // padded LDS row stride (bf16 elements): 64 + 8

typedef __attribute__((ext_vector_type(8)))  float  v8f;
typedef __attribute__((ext_vector_type(8)))  __bf16 v8bf;
typedef __attribute__((ext_vector_type(16))) __bf16 v16bf;
typedef __attribute__((ext_vector_type(4)))  int    v4i;
union BfFrag { v16bf v; v8bf h[2]; };

// ---- async global->LDS path (gfx1250), guarded so compilation never breaks
#if __has_builtin(__builtin_amdgcn_global_load_async_to_lds_b128) && \
    __has_builtin(__builtin_amdgcn_s_wait_asynccnt)
#define USE_ASYNC_LDS 1
#else
#define USE_ASYNC_LDS 0
#endif

// Builtin signature (from compiler diagnostic): param0 = int4* in AS1 (global),
// param1 = int4* in AS3 (LDS), then two immediate ints (offset, cpol).
using as1_v4i = __attribute__((address_space(1))) v4i;
using as3_v4i = __attribute__((address_space(3))) v4i;

static __device__ __forceinline__ as1_v4i* to_as1(const void* p) {
  return (as1_v4i*)(unsigned long long)p;  // global VA is identical
}
static __device__ __forceinline__ as3_v4i* to_as3(const void* p) {
  // low 32 bits of a generic LDS pointer are the LDS byte offset
  return (as3_v4i*)(unsigned)(unsigned long long)p;
}

// round-to-nearest-even f32 -> bf16, packed pair into one dword
static __device__ __forceinline__ unsigned f2bf_pack(float lo, float hi) {
  union { float f; unsigned u; } a, b;
  a.f = lo; b.f = hi;
  unsigned ra = (a.u + 0x7FFFu + ((a.u >> 16) & 1u)) >> 16;
  unsigned rb = (b.u + 0x7FFFu + ((b.u >> 16) & 1u)) >> 16;
  return ra | (rb << 16);
}

// ---------------------------------------------------------------------------
// Prep 1: alpha[k,h] = sigmoid(taus), cvec[h] = sum_k (1-alpha)*bias
// ---------------------------------------------------------------------------
__global__ void prep_alpha_kernel(const float* __restrict__ taus,
                                  const float* __restrict__ bias,
                                  float* __restrict__ alpha,
                                  float* __restrict__ cvec) {
  int h = blockIdx.x * blockDim.x + threadIdx.x;
  if (h >= H_DIM) return;
  float c = 0.f;
#pragma unroll
  for (int k = 0; k < K_BR; ++k) {
    float t = taus[k * H_DIM + h];
    float a = 1.f / (1.f + __expf(-t));
    alpha[k * H_DIM + h] = a;
    c += (1.f - a) * bias[k * H_DIM + h];
  }
  cvec[h] = c;
}

// ---------------------------------------------------------------------------
// Prep 2: fold (1-alpha) into weights, convert to bf16, fuse K branches:
//   Wbig[h, k*BR + i] = (1-alpha[k,h]) * W[k,h,i]   (bf16, N-major layout)
// ---------------------------------------------------------------------------
__global__ void prep_w_kernel(const float* __restrict__ W,
                              const float* __restrict__ alpha,
                              unsigned* __restrict__ Wbig_u32) {
  int kh = blockIdx.x * blockDim.x + threadIdx.x;  // 0 .. K_BR*H_DIM-1
  int k = kh >> 10;
  int h = kh & (H_DIM - 1);
  float s = 1.f - alpha[k * H_DIM + h];
  const float4* src = (const float4*)(W + ((size_t)k * H_DIM + h) * BR_DIM);
  unsigned* dst = Wbig_u32 + ((size_t)h * KD + (size_t)k * BR_DIM) / 2;
#pragma unroll 4
  for (int i = 0; i < BR_DIM / 4; ++i) {
    float4 f = src[i];
    dst[i * 2 + 0] = f2bf_pack(s * f.x, s * f.y);
    dst[i * 2 + 1] = f2bf_pack(s * f.z, s * f.w);
  }
}

// ---------------------------------------------------------------------------
// Prep 3: concat+convert activations once: Xbf[b, j] = bf16(x_full[b, j]).
// Converting once (instead of per N-block) kills the in-loop VALU packing and
// shrinks A to 32 MB bf16 -> the whole A matrix lives in the 192 MB L2.
// ---------------------------------------------------------------------------
__global__ void prep_x_kernel(const float* __restrict__ input_t,
                              const float* __restrict__ hidden,
                              unsigned* __restrict__ Xbf_u32) {
  unsigned t = blockIdx.x * blockDim.x + threadIdx.x;  // B_DIM*256 threads
  int b = t >> 8;
  int j = (t & 255) * 8;
  const float* src = (j < IN_DIM) ? input_t + (size_t)b * IN_DIM + j
                                  : hidden  + (size_t)b * H_DIM + (j - IN_DIM);
  unsigned* dst = Xbf_u32 + ((size_t)b * KD + j) / 2;
  float4 f0 = ((const float4*)src)[0];
  float4 f1 = ((const float4*)src)[1];
  dst[0] = f2bf_pack(f0.x, f0.y);
  dst[1] = f2bf_pack(f0.z, f0.w);
  dst[2] = f2bf_pack(f1.x, f1.y);
  dst[3] = f2bf_pack(f1.z, f1.w);
}

// ---------------------------------------------------------------------------
// Main fused GEMM + leaky-state + LIF threshold kernel.
//   out[b,h] = heaviside( 0.5*(gemm[b,h] + cvec[h] + sum_k alpha*bs[b,k,h]) - 1 )
// Double-buffered LDS; next tile staged via GLOBAL_LOAD_ASYNC_TO_LDS_B128
// while the current tile is consumed by v_wmma_f32_16x16x32_bf16.
// ---------------------------------------------------------------------------
__global__ __launch_bounds__(256) void gemm_spike_kernel(
    const unsigned short* __restrict__ Xbf,
    const unsigned short* __restrict__ Wbig,
    const float* __restrict__ bstates,
    const float* __restrict__ alpha,
    const float* __restrict__ cvec,
    float* __restrict__ out) {
  __shared__ unsigned short As[2][BM * LDSS];  // 128 x 64 bf16 (padded)
  __shared__ unsigned short Bs[2][BN * LDSS];  // 128(N) x 64(K) bf16 (padded)

  const int tid  = threadIdx.x;
  const int lane = tid & 31;
  const int wave = tid >> 5;
  const int wm   = wave & 3;   // wave tile row (4 x 32 rows)
  const int wn   = wave >> 2;  // wave tile col (2 x 64 cols)
  const int half = lane >> 4;
  const int l16  = lane & 15;

  const int tileM = blockIdx.x * BM;
  const int tileN = blockIdx.y * BN;

  v8f acc[2][4] = {};

  // staging assignment: 2 threads per row, 32 bf16 (64 B) each
  const int row  = tid >> 1;
  const int col  = (tid & 1) * 32;
  const unsigned short* aglob = Xbf  + (size_t)(tileM + row) * KD + col;
  const unsigned short* bglob = Wbig + (size_t)(tileN + row) * KD + col;

  auto stage = [&](int buf, int k0) {
    const unsigned short* ga = aglob + k0;
    const unsigned short* gb = bglob + k0;
    unsigned short* la = &As[buf][row * LDSS + col];
    unsigned short* lb = &Bs[buf][row * LDSS + col];
#if USE_ASYNC_LDS
#pragma unroll
    for (int v = 0; v < 4; ++v) {
      __builtin_amdgcn_global_load_async_to_lds_b128(to_as1(ga + v * 8),
                                                     to_as3(la + v * 8), 0, 0);
      __builtin_amdgcn_global_load_async_to_lds_b128(to_as1(gb + v * 8),
                                                     to_as3(lb + v * 8), 0, 0);
    }
#else
#pragma unroll
    for (int v = 0; v < 4; ++v) {
      ((uint4*)la)[v] = ((const uint4*)ga)[v];
      ((uint4*)lb)[v] = ((const uint4*)gb)[v];
    }
#endif
  };

  // prologue: fill buffer 0
  stage(0, 0);
#if USE_ASYNC_LDS
  __builtin_amdgcn_s_wait_asynccnt(0);
#endif
  __syncthreads();

  const int abase = half * 8;   // A frag: lanes0-15 K0-7/16-23, lanes16-31 K8-15/24-31
  const int kb    = half * 16;  // unused; B frag uses A's order: lanes0-15 K0-7/16-23, lanes16-31 K8-15/24-31

  int buf = 0;
  for (int k0 = 0; k0 < KD; k0 += BK, buf ^= 1) {
    // stage next tile into the other buffer (async, overlapped with WMMA)
    if (k0 + BK < KD) stage(buf ^ 1, k0 + BK);

    // consume current buffer: 2 k-steps of 32, 8 WMMAs each
#pragma unroll
    for (int ks = 0; ks < 2; ++ks) {
      const int ko = ks * 32;
      BfFrag afrag[2], bfrag[4];
#pragma unroll
      for (int mi = 0; mi < 2; ++mi) {
        const unsigned short* p =
            &As[buf][(wm * 32 + mi * 16 + l16) * LDSS + ko];
        afrag[mi].h[0] = *(const v8bf*)(const void*)(p + abase);
        afrag[mi].h[1] = *(const v8bf*)(const void*)(p + 16 + abase);
      }
#pragma unroll
      for (int ni = 0; ni < 4; ++ni) {
        const unsigned short* p =
            &Bs[buf][(wn * 64 + ni * 16 + l16) * LDSS + ko];
        bfrag[ni].h[0] = *(const v8bf*)(const void*)(p + half * 8);
        bfrag[ni].h[1] = *(const v8bf*)(const void*)(p + 16 + half * 8);
      }
#pragma unroll
      for (int mi = 0; mi < 2; ++mi)
#pragma unroll
        for (int ni = 0; ni < 4; ++ni)
          acc[mi][ni] = __builtin_amdgcn_wmma_f32_16x16x32_bf16(
              false, afrag[mi].v, false, bfrag[ni].v,
              (short)0, acc[mi][ni], false, false);
    }

#if USE_ASYNC_LDS
    __builtin_amdgcn_s_wait_asynccnt(0);
#endif
    __syncthreads();
  }

  // ---- epilogue: + cvec + alpha*branch_states, /tau, threshold ----
#pragma unroll
  for (int ni = 0; ni < 4; ++ni) {
    const int n = tileN + wn * 64 + ni * 16 + l16;
    const float a0 = alpha[0 * H_DIM + n];
    const float a1 = alpha[1 * H_DIM + n];
    const float a2 = alpha[2 * H_DIM + n];
    const float a3 = alpha[3 * H_DIM + n];
    const float cn = cvec[n];
#pragma unroll
    for (int mi = 0; mi < 2; ++mi) {
#pragma unroll
      for (int r = 0; r < 8; ++r) {
        const int m = tileM + wm * 32 + mi * 16 + half * 8 + r;
        const float* bs = bstates + (size_t)m * (K_BR * H_DIM) + n;
        const float s = a0 * bs[0 * H_DIM] + a1 * bs[1 * H_DIM] +
                        a2 * bs[2 * H_DIM] + a3 * bs[3 * H_DIM];
        const float tot = acc[mi][ni][r] + cn + s;
        out[(size_t)m * H_DIM + n] = (0.5f * tot - 1.0f >= 0.f) ? 1.f : 0.f;
      }
    }
  }
}

// ---------------------------------------------------------------------------
extern "C" void kernel_launch(void* const* d_in, const int* in_sizes, int n_in,
                              void* d_out, int out_size, void* d_ws, size_t ws_size,
                              hipStream_t stream) {
  const float* input_t = (const float*)d_in[0];  // [8192,1024]
  const float* hidden  = (const float*)d_in[1];  // [8192,1024]
  const float* bstates = (const float*)d_in[2];  // [8192,4,1024]
  const float* W       = (const float*)d_in[3];  // [4,1024,512]
  const float* bias    = (const float*)d_in[4];  // [4,1024]
  const float* taus    = (const float*)d_in[5];  // [4,1024]
  float* out = (float*)d_out;                    // [8192,1024]

  // workspace: Xbf bf16 [B][KD] (32 MB) | Wbig bf16 [H][KD] (4 MB) | alpha | cvec
  char* ws = (char*)d_ws;
  unsigned short* Xbf  = (unsigned short*)ws;
  unsigned short* Wbig = (unsigned short*)(ws + (size_t)B_DIM * KD * 2);
  float* alpha = (float*)(ws + (size_t)B_DIM * KD * 2 + (size_t)H_DIM * KD * 2);
  float* cvec  = alpha + K_BR * H_DIM;

  prep_alpha_kernel<<<H_DIM / 256, 256, 0, stream>>>(taus, bias, alpha, cvec);
  prep_x_kernel<<<B_DIM, 256, 0, stream>>>(input_t, hidden, (unsigned*)Xbf);
  prep_w_kernel<<<(K_BR * H_DIM) / 256, 256, 0, stream>>>(
      W, alpha, (unsigned*)Wbig);

  dim3 grid(B_DIM / BM, H_DIM / BN);  // 64 x 8
  gemm_spike_kernel<<<grid, 256, 0, stream>>>(
      Xbf, Wbig, bstates, alpha, cvec, out);
}
